// filter_model_5660766896493
// MI455X (gfx1250) — compile-verified, matrix operand order checked by reading, not run on hardware
//
#include <hip/hip_runtime.h>
#include <hip/hip_bf16.h>
#include <stdint.h>

typedef _Float16 v16h __attribute__((ext_vector_type(16)));
typedef _Float16 v8h  __attribute__((ext_vector_type(8)));
typedef float    v8f  __attribute__((ext_vector_type(8)));

#define NB 8       // batch
#define NA 720     // angles
#define ND 1024    // detector pixels
#define NH 512
#define NW 512

// ---------------------------------------------------------------------------
// CDNA5 async memory->LDS copy (ASYNCcnt-tracked), inline asm for portability
// across ROCm 7.2 / amdgpu-toolchain builtin arity differences.
// VDST = LDS byte offset (low 32 bits of a generic shared pointer),
// VADDR = 64-bit global address.
// ---------------------------------------------------------------------------
__device__ __forceinline__ void async_load_b128(uint32_t lds_off, const float* gaddr) {
    asm volatile("global_load_async_to_lds_b128 %0, %1, off"
                 :: "v"(lds_off), "v"(gaddr)
                 : "memory");
}
__device__ __forceinline__ void wait_async0() {
    asm volatile("s_wait_asynccnt 0x0" ::: "memory");
}

// ---------------------------------------------------------------------------
// Kernel 1: trig table (cos/sin per angle) + spatial ramp-filter kernel
//   h[n] = (1/D) * sum_f w[f] * cos(2*pi*f*n/D)   (w real & even -> h real, even)
// Integer phase (f*n) & 1023 keeps the cosine argument in [0, 2*pi).
// ---------------------------------------------------------------------------
__global__ __launch_bounds__(1024) void prep_kernel(const float* __restrict__ fw,
                                                    float* __restrict__ trig,
                                                    float* __restrict__ kf) {
    const int n = threadIdx.x;                       // 0..1023
    const float twopi_over_d = 6.28318530717958647692f / 1024.0f;
    float acc = 0.0f;
    for (int f = 0; f < ND; ++f) {
        int ph = (f * n) & (ND - 1);
        acc = fmaf(fw[f], cosf((float)ph * twopi_over_d), acc);
    }
    kf[n] = acc * (1.0f / (float)ND);

    if (n < NA) {
        float th = 3.14159265358979323846f * (float)n / (float)NA;
        trig[2 * n + 0] = cosf(th);
        trig[2 * n + 1] = sinf(th);
    }
}

// ---------------------------------------------------------------------------
// Kernel 2: materialize circulant filter matrix in f16.
//   Kmat[i][j] = h[(i-j) & 1023]   (symmetric: h is even)
// ---------------------------------------------------------------------------
__global__ __launch_bounds__(256) void buildk_kernel(const float* __restrict__ kf,
                                                     _Float16* __restrict__ Kmat) {
    int idx = blockIdx.x * 256 + threadIdx.x;        // 0 .. 1M-1
    int i = idx >> 10;
    int j = idx & (ND - 1);
    Kmat[idx] = (_Float16)kf[(i - j) & (ND - 1)];
}

// ---------------------------------------------------------------------------
// Kernel 3: filtered = X[5760x1024] * K[1024x1024] with v_wmma_f32_16x16x32_f16.
// One wave -> one 16(M) x 64(N) strip (4 accumulators, A fragment reused 4x).
// A loaded from f32 x (converted inline); B loaded as contiguous v8h from the
// symmetric K (row n over k).
// CDNA5 16-bit A layout: row = lane&15; lane>>4 selects K-halves
//   halves 0..7  -> K = kb + (lane>>4)*8 + 0..7
//   halves 8..15 -> K = kb + 16 + (lane>>4)*8 + 0..7
// ---------------------------------------------------------------------------
__global__ __launch_bounds__(256) void gemm_kernel(const float* __restrict__ x,
                                                   const _Float16* __restrict__ Kmat,
                                                   float* __restrict__ filtered) {
    const int lane  = threadIdx.x & 31;
    const int wave  = threadIdx.x >> 5;
    const int wgid  = blockIdx.x * 8 + wave;         // 0..5759
    const int mTile = wgid >> 4;                     // 0..359
    const int nStrip = wgid & 15;                    // 0..15

    const int lrow = lane & 15;
    const int kg   = (lane >> 4) * 8;

    const float* arow = x + (size_t)(mTile * 16 + lrow) * ND;
    const _Float16* brow0 = Kmat + (size_t)(nStrip * 64 +  0 + lrow) * ND;
    const _Float16* brow1 = Kmat + (size_t)(nStrip * 64 + 16 + lrow) * ND;
    const _Float16* brow2 = Kmat + (size_t)(nStrip * 64 + 32 + lrow) * ND;
    const _Float16* brow3 = Kmat + (size_t)(nStrip * 64 + 48 + lrow) * ND;
    const _Float16* brow[4] = {brow0, brow1, brow2, brow3};

    v8f acc[4];
#pragma unroll
    for (int sub = 0; sub < 4; ++sub) acc[sub] = (v8f){0.f,0.f,0.f,0.f,0.f,0.f,0.f,0.f};

    for (int kk = 0; kk < ND / 32; ++kk) {
        const int kb = kk * 32;
        float4 a0 = *(const float4*)(arow + kb + kg);
        float4 a1 = *(const float4*)(arow + kb + kg + 4);
        float4 a2 = *(const float4*)(arow + kb + 16 + kg);
        float4 a3 = *(const float4*)(arow + kb + 16 + kg + 4);
        v16h a;
        a[0]  = (_Float16)a0.x; a[1]  = (_Float16)a0.y; a[2]  = (_Float16)a0.z; a[3]  = (_Float16)a0.w;
        a[4]  = (_Float16)a1.x; a[5]  = (_Float16)a1.y; a[6]  = (_Float16)a1.z; a[7]  = (_Float16)a1.w;
        a[8]  = (_Float16)a2.x; a[9]  = (_Float16)a2.y; a[10] = (_Float16)a2.z; a[11] = (_Float16)a2.w;
        a[12] = (_Float16)a3.x; a[13] = (_Float16)a3.y; a[14] = (_Float16)a3.z; a[15] = (_Float16)a3.w;

#pragma unroll
        for (int sub = 0; sub < 4; ++sub) {
            v8h blo = *(const v8h*)(brow[sub] + kb + kg);
            v8h bhi = *(const v8h*)(brow[sub] + kb + 16 + kg);
            v16h b;
#pragma unroll
            for (int i2 = 0; i2 < 8; ++i2) { b[i2] = blo[i2]; b[8 + i2] = bhi[i2]; }
            acc[sub] = __builtin_amdgcn_wmma_f32_16x16x32_f16(
                false, a, false, b, (short)0, acc[sub], false, false);
        }
    }

    // D layout: n = lane&15 (+ strip), m = (lane>>4)*8 + i (+ tile)
#pragma unroll
    for (int sub = 0; sub < 4; ++sub) {
#pragma unroll
        for (int i = 0; i < 8; ++i) {
            int m = mTile * 16 + (lane >> 4) * 8 + i;
            int n = nStrip * 64 + sub * 16 + lrow;
            filtered[(size_t)m * ND + n] = acc[sub][i];
        }
    }
}

// ---------------------------------------------------------------------------
// Kernel 4: backprojection. Block = 256 threads = 32x32 pixel tile
// (tx = lane -> x, wave -> y rows {w, w+8, w+16, w+24}). Per angle, the
// 8x1024-float sinogram slab is copied memory->LDS with CDNA5 async loads
// (global_load_async_to_lds_b128, ASYNCcnt), double buffered (64 KB); the
// copy for angle a+1 is in flight while angle a is interpolated out of LDS.
// ---------------------------------------------------------------------------
__global__ __launch_bounds__(256) void bp_kernel(const float* __restrict__ sino,
                                                 const float* __restrict__ trig,
                                                 float* __restrict__ out) {
    __shared__ float smem[2][NB * ND];               // 2 * 32 KB

    const int tid = threadIdx.x;
    const int tx  = tid & 31;
    const int wid = tid >> 5;
    const int bx  = blockIdx.x & 15;                 // 16 x-tiles
    const int by  = blockIdx.x >> 4;                 // 16 y-tiles

    // LDS byte offset of the staging buffers (aperture rule: low 32 bits of a
    // generic shared pointer == LDS byte address).
    const uint32_t smem_base = (uint32_t)(uintptr_t)(&smem[0][0]);

    const float X = (float)(bx * 32 + tx) - 255.5f;
    float Y[4];
#pragma unroll
    for (int p = 0; p < 4; ++p) Y[p] = (float)(by * 32 + wid + 8 * p) - 255.5f;

    float acc[4][8];
#pragma unroll
    for (int p = 0; p < 4; ++p)
#pragma unroll
        for (int b = 0; b < 8; ++b) acc[p][b] = 0.0f;

    // prologue: kick off async copy of angle 0 into buffer 0
#pragma unroll
    for (int b = 0; b < 8; ++b)
        async_load_b128(smem_base + (uint32_t)(b * ND + tid * 4) * 4u,
                        sino + (size_t)(b * NA) * ND + tid * 4);

    for (int a = 0; a < NA; ++a) {
        wait_async0();                               // my copies into buf (a&1) done
        __syncthreads();                             // everyone's done; other buf free
        const int cur = a & 1;

        if (a + 1 < NA) {                            // async-copy next angle -> other buf
            const uint32_t base = smem_base + (uint32_t)(cur ^ 1) * (NB * ND * 4u);
#pragma unroll
            for (int b = 0; b < 8; ++b)
                async_load_b128(base + (uint32_t)(b * ND + tid * 4) * 4u,
                                sino + (size_t)(b * NA + a + 1) * ND + tid * 4);
        }

        const float c = trig[2 * a + 0];
        const float s = trig[2 * a + 1];
        const float* L = &smem[cur][0];

#pragma unroll
        for (int p = 0; p < 4; ++p) {
            float t  = fmaf(X, c, fmaf(Y[p], s, 511.5f));
            float fi = floorf(t);
            int   i0 = (int)fi;
            float w  = t - fi;
            int   i1 = i0 + 1;
            float w0 = (i0 >= 0 && i0 < ND) ? (1.0f - w) : 0.0f;
            float w1 = (i1 >= 0 && i1 < ND) ? w : 0.0f;
            int i0c = i0 < 0 ? 0 : (i0 > ND - 1 ? ND - 1 : i0);
            int i1c = i1 < 0 ? 0 : (i1 > ND - 1 ? ND - 1 : i1);
#pragma unroll
            for (int b = 0; b < 8; ++b) {
                acc[p][b] = fmaf(L[b * ND + i0c], w0,
                            fmaf(L[b * ND + i1c], w1, acc[p][b]));
            }
        }
    }

    const float scale = 3.14159265358979323846f / (float)NA;
#pragma unroll
    for (int p = 0; p < 4; ++p) {
        int y  = by * 32 + wid + 8 * p;
        int xg = bx * 32 + tx;
#pragma unroll
        for (int b = 0; b < 8; ++b)
            out[(size_t)b * (NH * NW) + (size_t)y * NW + xg] = acc[p][b] * scale;
    }
}

// ---------------------------------------------------------------------------
extern "C" void kernel_launch(void* const* d_in, const int* in_sizes, int n_in,
                              void* d_out, int out_size, void* d_ws, size_t ws_size,
                              hipStream_t stream) {
    const float* x  = (const float*)d_in[0];   // [8, 720, 1024] f32
    const float* fw = (const float*)d_in[1];   // [1024] f32
    float* out = (float*)d_out;                // [8, 512, 512] f32

    char* ws = (char*)d_ws;
    float*     trig     = (float*)ws;                          // 1440 floats
    float*     kf       = (float*)(ws + 8192);                 // 1024 floats
    _Float16*  Kmat     = (_Float16*)(ws + 16384);             // 2 MB
    float*     filtered = (float*)(ws + 16384 + (2u << 20));   // 5760*1024 f32 = 23.6 MB

    prep_kernel  <<<1, 1024, 0, stream>>>(fw, trig, kf);
    buildk_kernel<<<(ND * ND) / 256, 256, 0, stream>>>(kf, Kmat);
    gemm_kernel  <<<720, 256, 0, stream>>>(x, Kmat, filtered);
    bp_kernel    <<<256, 256, 0, stream>>>(filtered, trig, out);
}
